// HieraMaskUnitAttention_137438954353
// MI455X (gfx1250) — matrix operand order repaired and checked by reading, not compiled, not run on hardware
//
#include <hip/hip_runtime.h>
#include <hip/hip_bf16.h>

// ---------------------------------------------------------------------------
// Hiera mask-unit attention for MI455X (gfx1250), wave32 + WMMA bf16 pipeline.
//   K1: qkv = X @ Wqkv^T + b           (bf16 out to ws)
//   K2: windowed attention per (b,h,w) (bf16 out to ws, hiera layout)
//   K3: out = A @ Wproj^T + b          (f32 out to d_out, A tile via TDM)
// ---------------------------------------------------------------------------

typedef __attribute__((ext_vector_type(16))) __bf16 v16bf;
typedef __attribute__((ext_vector_type(8)))  float  v8f;
typedef __attribute__((ext_vector_type(4)))  unsigned int v4u;
typedef __attribute__((ext_vector_type(4)))  int  v4i;
typedef __attribute__((ext_vector_type(8)))  int  v8i_t;

union Frag { unsigned int u[8]; v16bf v; };

#define N_B    32
#define N_S    4096
#define CIN    192
#define COUT   384
#define NHEAD  4
#define HD     96
#define NWIN   16
#define WINP   256      // tokens per window
#define SQ     64       // pooled queries per window
#define QKV_N  1152     // 3*COUT
#define ATT_SCALE 0.10206207261596577f   // 96^-0.5

static __device__ __forceinline__ unsigned short f2bf(float f) {
    union { float f; unsigned int u; } c; c.f = f;
    unsigned int u = c.u;
    u += 0x7FFFu + ((u >> 16) & 1u);       // round-to-nearest-even
    return (unsigned short)(u >> 16);
}
static __device__ __forceinline__ float bf2f(unsigned short h) {
    union { float f; unsigned int u; } c; c.u = ((unsigned int)h) << 16;
    return c.f;
}

static __device__ __forceinline__ v8f wmma_bf16(const Frag& a, const Frag& b, v8f c) {
    return __builtin_amdgcn_wmma_f32_16x16x32_bf16(
        /*neg_a=*/false, a.v, /*neg_b=*/false, b.v,
        /*c_mod=*/(short)0, c, /*reuse_a=*/false, /*reuse_b=*/false);
}

// ---------------------------------------------------------------------------
// Tensor Data Mover: 2D bf16 tile (tile_d1 rows x tile_d0 elems) from global
// to LDS.  D# per CDNA5 ISA: group0 = {count,
// lds_addr, global_addr, type=2}; group1 = {data_size=2B, tensor dims, tile
// dims, dim0 stride}; groups 2/3 zero (<=2D tensor).
// ---------------------------------------------------------------------------
static __device__ __forceinline__ void tdm_load_2d_bf16(
    unsigned int lds_off, const void* gptr,
    unsigned int tensor_d0, unsigned int tensor_d1,
    unsigned int tile_d0, unsigned int tile_d1,
    unsigned long long stride0_elems) {
    unsigned long long ga = (unsigned long long)(uintptr_t)gptr;
    v4u g0;
    g0.x = 1u;                                            // count=1 (valid, user)
    g0.y = lds_off;                                       // lds_addr [63:32]
    g0.z = (unsigned int)(ga & 0xffffffffu);              // global_addr [95:64]
    g0.w = (unsigned int)((ga >> 32) & 0x01ffffffu)       // global_addr [120:96]
         | (2u << 30);                                    // type=2 ("image")
    v8i_t g1;
    g1[0] = (int)(1u << 16);                              // data_size=1 -> 2 bytes
    g1[1] = (int)((tensor_d0 & 0xffffu) << 16);           // tensor_dim0 lo16
    g1[2] = (int)((tensor_d0 >> 16) | ((tensor_d1 & 0xffffu) << 16));
    g1[3] = (int)((tensor_d1 >> 16) | ((tile_d0 & 0xffffu) << 16));
    g1[4] = (int)(tile_d1 & 0xffffu);                     // tile_dim1 (tile_dim2=0)
    g1[5] = (int)(stride0_elems & 0xffffffffu);           // tensor_dim0_stride lo32
    g1[6] = (int)((stride0_elems >> 32) & 0xffffu);       // stride hi16 (dim1 stride=0)
    g1[7] = 0;
    v4i gz = {0, 0, 0, 0};
#if defined(__clang_major__) && (__clang_major__ >= 23)
    v8i_t gz8 = {0, 0, 0, 0, 0, 0, 0, 0};
    __builtin_amdgcn_tensor_load_to_lds(g0, g1, gz, gz, gz8, 0);
#else
    __builtin_amdgcn_tensor_load_to_lds(g0, g1, gz, gz, 0);
#endif
}

// ---------------------------------------------------------------------------
// Kernel 1: QKV projection.  C[m,n] = sum_k X[m,k] * W[n,k] + bias[n]
// M = 131072, K = 192, N = 1152.  Block tile 128x64, 8 waves (wave = 16 rows).
// ---------------------------------------------------------------------------
__global__ __launch_bounds__(256) void qkv_gemm_k(
    const float* __restrict__ X, const float* __restrict__ W,
    const float* __restrict__ bias, unsigned short* __restrict__ Y) {
    const int n0 = blockIdx.x * 64;
    const int m0 = blockIdx.y * 128;
    __shared__ __align__(16) unsigned short As[128 * 36];  // padded stride 36
    __shared__ __align__(16) unsigned short Bs[64 * 36];

    const int tid  = threadIdx.x;
    const int lane = tid & 31, wid = tid >> 5;
    const int g = lane >> 4, ln = lane & 15;

    v8f acc[4];
    const v8f vz = {0.f,0.f,0.f,0.f,0.f,0.f,0.f,0.f};
#pragma unroll
    for (int nt = 0; nt < 4; ++nt) acc[nt] = vz;

    for (int kb = 0; kb < CIN; kb += 32) {
#pragma unroll
        for (int i = 0; i < 16; ++i) {            // A tile: 128x32
            int e = tid + i * 256, r = e >> 5, k = e & 31;
            As[r * 36 + k] = f2bf(X[(size_t)(m0 + r) * CIN + kb + k]);
        }
#pragma unroll
        for (int i = 0; i < 8; ++i) {             // B tile: 64x32
            int e = tid + i * 256, r = e >> 5, k = e & 31;
            Bs[r * 36 + k] = f2bf(W[(size_t)(n0 + r) * CIN + kb + k]);
        }
        __syncthreads();
        if (kb + 32 < CIN)                        // -> global_prefetch_b8
            __builtin_prefetch(&X[(size_t)(m0 + (tid >> 1)) * CIN + kb + 32], 0, 1);

        Frag a;
#pragma unroll
        for (int v = 0; v < 8; ++v) {             // A frag: M=lane row, K pairs
            int k = 2 * v + 16 * g;
            a.u[v] = *(const unsigned int*)&As[(wid * 16 + ln) * 36 + k];
        }
#pragma unroll
        for (int nt = 0; nt < 4; ++nt) {
            Frag b;
#pragma unroll
            for (int v = 0; v < 8; ++v) {         // B frag: N=lane col, K=2v+16g
                int k = 2 * v + 16 * g;
                b.u[v] = *(const unsigned int*)&Bs[(nt * 16 + ln) * 36 + k];
            }
            acc[nt] = wmma_bf16(a, b, acc[nt]);
        }
        __syncthreads();
    }
#pragma unroll
    for (int nt = 0; nt < 4; ++nt) {
        int n = n0 + nt * 16 + ln;
        float bv = bias[n];
#pragma unroll
        for (int v = 0; v < 8; ++v) {
            int m = m0 + wid * 16 + v + 8 * g;    // C layout: M = v + 8g
            Y[(size_t)m * QKV_N + n] = f2bf(acc[nt][v] + bv);
        }
    }
}

// ---------------------------------------------------------------------------
// Kernel 2: windowed attention for one (b, h, w).  4 waves / 128 threads.
// Token p of window w is sequence index p*16 + w.  Pooled q_j = max_i q[i*64+j].
// ---------------------------------------------------------------------------
__global__ __launch_bounds__(128) void mask_attn_k(
    const unsigned short* __restrict__ QKV, unsigned short* __restrict__ O) {
    const int h = blockIdx.x >> 4, w = blockIdx.x & 15;
    const int b = blockIdx.y;
    __shared__ __align__(16) unsigned short Qs[SQ * 100];   // pooled q * scale
    __shared__ __align__(16) unsigned short Vs[HD * 36];    // V chunk, [d][tok]
    __shared__ __align__(16) unsigned short Ps[4][16 * 36]; // per-wave P staging

    const int tid = threadIdx.x;
    const int lane = tid & 31, wid = tid >> 5;
    const int g = lane >> 4, ln = lane & 15;
    const size_t rowbase = (size_t)b * N_S * QKV_N;
    const int qoff = h * HD, koff = COUT + h * HD, voff = 2 * COUT + h * HD;

    // --- Q-stride max pool (pre-scaled) into LDS ---
    for (int e = tid; e < SQ * HD; e += 128) {
        int j = e / HD, d = e % HD;
        float m = -3.402823e38f;
#pragma unroll
        for (int i4 = 0; i4 < 4; ++i4) {
            int p = i4 * 64 + j;
            m = fmaxf(m, bf2f(QKV[rowbase + (size_t)(p * 16 + w) * QKV_N + qoff + d]));
        }
        Qs[j * 100 + d] = f2bf(m * ATT_SCALE);
    }
    __syncthreads();

    // --- scores S = (Q*scale) @ K^T : wave rows [wid*16, +16), all 256 keys ---
    Frag aq[3];
#pragma unroll
    for (int dc = 0; dc < 3; ++dc)
#pragma unroll
        for (int v = 0; v < 8; ++v) {
            int k = dc * 32 + 2 * v + 16 * g;
            aq[dc].u[v] = *(const unsigned int*)&Qs[(wid * 16 + ln) * 100 + k];
        }
    v8f sacc[16];
#pragma unroll
    for (int kt = 0; kt < 16; ++kt) {
        v8f c = {0.f,0.f,0.f,0.f,0.f,0.f,0.f,0.f};
        int p = kt * 16 + ln;                         // key token (B-frag column)
#pragma unroll
        for (int dc = 0; dc < 3; ++dc) {
            Frag bk;
#pragma unroll
            for (int v = 0; v < 8; ++v) {             // K[p][d,d+1] contiguous pair
                int d = dc * 32 + 2 * v + 16 * g;
                bk.u[v] = *(const unsigned int*)
                    &QKV[rowbase + (size_t)(p * 16 + w) * QKV_N + koff + d];
            }
            c = wmma_bf16(aq[dc], bk, c);
        }
        sacc[kt] = c;
    }

    // --- softmax: row m = v + 8g lives in one 16-lane half; shfl-xor reduce ---
#pragma unroll
    for (int v = 0; v < 8; ++v) {
        float m = sacc[0][v];
#pragma unroll
        for (int kt = 1; kt < 16; ++kt) m = fmaxf(m, sacc[kt][v]);
#pragma unroll
        for (int msk = 1; msk < 16; msk <<= 1) m = fmaxf(m, __shfl_xor(m, msk, 32));
        float s = 0.f;
#pragma unroll
        for (int kt = 0; kt < 16; ++kt) {
            float e = __expf(sacc[kt][v] - m);
            sacc[kt][v] = e; s += e;
        }
#pragma unroll
        for (int msk = 1; msk < 16; msk <<= 1) s += __shfl_xor(s, msk, 32);
        float inv = 1.f / s;
#pragma unroll
        for (int kt = 0; kt < 16; ++kt) sacc[kt][v] *= inv;   // P = softmax rows
    }

    // --- O = P @ V over 8 chunks of 32 tokens ---
    v8f oacc[6];
    const v8f vz = {0.f,0.f,0.f,0.f,0.f,0.f,0.f,0.f};
#pragma unroll
    for (int nt = 0; nt < 6; ++nt) oacc[nt] = vz;
#pragma unroll
    for (int c = 0; c < 8; ++c) {
        __syncthreads();                              // protect Vs reuse
        for (int e = tid; e < 32 * HD; e += 128) {    // V chunk transposed [d][tok]
            int tk = e / HD, d = e % HD;
            Vs[d * 36 + tk] =
                QKV[rowbase + (size_t)((c * 32 + tk) * 16 + w) * QKV_N + voff + d];
        }
        __syncthreads();
        // stage this wave's P chunk (C layout -> A layout via LDS)
#pragma unroll
        for (int t = 0; t < 2; ++t)
#pragma unroll
            for (int v = 0; v < 8; ++v)
                Ps[wid][(v + 8 * g) * 36 + t * 16 + ln] = f2bf(sacc[2 * c + t][v]);
        Frag ap;
#pragma unroll
        for (int v = 0; v < 8; ++v) {                 // A frag of P (per-wave, in-order LDS)
            int k = 2 * v + 16 * g;
            ap.u[v] = *(const unsigned int*)&Ps[wid][ln * 36 + k];
        }
#pragma unroll
        for (int nt = 0; nt < 6; ++nt) {
            Frag bv;
#pragma unroll
            for (int v = 0; v < 8; ++v) {             // V^T pair: tokens 2v+16g, +1 at d
                int k = 2 * v + 16 * g;
                bv.u[v] = *(const unsigned int*)&Vs[(nt * 16 + ln) * 36 + k];
            }
            oacc[nt] = wmma_bf16(ap, bv, oacc[nt]);
        }
    }

    // --- store O in hiera layout: seq = j*16 + w, chan = h*96 + d ---
#pragma unroll
    for (int nt = 0; nt < 6; ++nt)
#pragma unroll
        for (int v = 0; v < 8; ++v) {
            int j = wid * 16 + v + 8 * g;
            int d = nt * 16 + ln;
            O[(size_t)(b * 1024 + j * 16 + w) * COUT + h * HD + d] = f2bf(oacc[nt][v]);
        }
}

// ---------------------------------------------------------------------------
// Kernel 3: output projection.  M = 32768, K = 384, N = 384, f32 out.
// A tile (bf16, 128x32) is DMA'd into LDS by the Tensor Data Mover.
// ---------------------------------------------------------------------------
__global__ __launch_bounds__(256) void proj_gemm_k(
    const unsigned short* __restrict__ A, const float* __restrict__ W,
    const float* __restrict__ bias, float* __restrict__ Y) {
    const int n0 = blockIdx.x * 64;
    const int m0 = blockIdx.y * 128;
    __shared__ __align__(16) unsigned short As[128 * 32];  // TDM tile, stride 32
    __shared__ __align__(16) unsigned short Bs[64 * 36];

    const int tid = threadIdx.x;
    const int lane = tid & 31, wid = tid >> 5;
    const int g = lane >> 4, ln = lane & 15;
    const unsigned int as_lds = (unsigned int)(uintptr_t)&As[0];

    v8f acc[4];
    const v8f vz = {0.f,0.f,0.f,0.f,0.f,0.f,0.f,0.f};
#pragma unroll
    for (int nt = 0; nt < 4; ++nt) acc[nt] = vz;

    for (int kb = 0; kb < COUT; kb += 32) {
        if (wid == 0) {                           // TDM: A tile 128x32 bf16 -> LDS
            tdm_load_2d_bf16(as_lds, &A[(size_t)m0 * COUT + kb],
                             /*tensor_d0=*/COUT, /*tensor_d1=*/N_B * 1024,
                             /*tile_d0=*/32, /*tile_d1=*/128,
                             /*stride0=*/COUT);
            __builtin_amdgcn_s_wait_tensorcnt(0);
        }
#pragma unroll
        for (int i = 0; i < 8; ++i) {             // B tile: 64x32 (f32 -> bf16)
            int e = tid + i * 256, r = e >> 5, k = e & 31;
            Bs[r * 36 + k] = f2bf(W[(size_t)(n0 + r) * COUT + kb + k]);
        }
        __syncthreads();
        if (kb + 32 < COUT)
            __builtin_prefetch(&W[(size_t)(n0 + (tid >> 2)) * COUT + kb + 32], 0, 1);

        Frag a;
#pragma unroll
        for (int v = 0; v < 8; ++v) {
            int k = 2 * v + 16 * g;
            a.u[v] = *(const unsigned int*)&As[(wid * 16 + ln) * 32 + k];
        }
#pragma unroll
        for (int nt = 0; nt < 4; ++nt) {
            Frag b;
#pragma unroll
            for (int v = 0; v < 8; ++v) {
                int k = 2 * v + 16 * g;
                b.u[v] = *(const unsigned int*)&Bs[(nt * 16 + ln) * 36 + k];
            }
            acc[nt] = wmma_bf16(a, b, acc[nt]);
        }
        __syncthreads();
    }
#pragma unroll
    for (int nt = 0; nt < 4; ++nt) {
        int n = n0 + nt * 16 + ln;
        float bv = bias[n];
#pragma unroll
        for (int v = 0; v < 8; ++v) {
            int m = m0 + wid * 16 + v + 8 * g;
            Y[(size_t)m * COUT + n] = acc[nt][v] + bv;
        }
    }
}

// ---------------------------------------------------------------------------
extern "C" void kernel_launch(void* const* d_in, const int* in_sizes, int n_in,
                              void* d_out, int out_size, void* d_ws, size_t ws_size,
                              hipStream_t stream) {
    const float* X      = (const float*)d_in[0];  // [32, 4096, 192]
    const float* qkv_w  = (const float*)d_in[1];  // [1152, 192]
    const float* qkv_b  = (const float*)d_in[2];  // [1152]
    const float* proj_w = (const float*)d_in[3];  // [384, 384]
    const float* proj_b = (const float*)d_in[4];  // [384]
    float* out = (float*)d_out;                   // [32, 1024, 384]

    unsigned short* qkv_ws  = (unsigned short*)d_ws;                       // [131072,1152] bf16
    unsigned short* attn_ws = qkv_ws + (size_t)N_B * N_S * QKV_N;          // [32768,384]  bf16

    // K1: QKV GEMM  (grid: 18 n-tiles x 1024 m-tiles)
    qkv_gemm_k<<<dim3(QKV_N / 64, (N_B * N_S) / 128), 256, 0, stream>>>(
        X, qkv_w, qkv_b, qkv_ws);
    // K2: attention (grid: (h,w) x b = 64 x 32)
    mask_attn_k<<<dim3(NHEAD * NWIN, N_B), 128, 0, stream>>>(qkv_ws, attn_ws);
    // K3: projection (grid: 6 n-tiles x 256 m-tiles)
    proj_gemm_k<<<dim3(COUT / 64, (N_B * 1024) / 128), 256, 0, stream>>>(
        attn_ws, proj_w, proj_b, out);
}
